// StateDecompressor_37074157699817
// MI455X (gfx1250) — matrix operand order repaired and checked by reading, not compiled, not run on hardware
//
#include <hip/hip_runtime.h>
#include <hip/hip_bf16.h>

// CDNA5 / gfx1250 persistent autoregressive LSTM decoder.
// All GEMMs via v_wmma_f32_16x16x32_bf16; c-state lives in accumulator-layout
// registers; activations in LDS; packed bf16 weights streamed from L2.

typedef __attribute__((ext_vector_type(16))) __bf16 v16bf;
typedef __attribute__((ext_vector_type(8)))  float  v8f;

union FragU { uint4 q[2]; v16bf v; };

__device__ __forceinline__ unsigned short f2bf(float f) {
  unsigned u = __float_as_uint(f);
  unsigned r = ((u >> 16) & 1u) + 0x7FFFu;   // round-to-nearest-even
  return (unsigned short)((u + r) >> 16);
}

__device__ __forceinline__ float sigm(float x) { return 1.0f / (1.0f + __expf(-x)); }

// A-fragment (16x32 bf16) from row-major bf16 LDS tile, per ISA 7.12.2 layout:
// lanes 0-15: K{0..7,16..23}; lanes 16-31: K{8..15,24..31} for row m=lane&15.
__device__ __forceinline__ v16bf load_a_frag(const unsigned short* Alds, int ldA,
                                             int kt, int lane) {
  const int m = lane & 15;
  const uint4* p = (const uint4*)((const char*)Alds + m * (ldA * 2) + kt * 64);
  FragU f;
  if (lane < 16) { f.q[0] = p[0]; f.q[1] = p[2]; }
  else           { f.q[0] = p[1]; f.q[1] = p[3]; }
  return f.v;
}

// B-fragment: pre-swizzled in global memory, 32 contiguous bytes per lane.
__device__ __forceinline__ v16bf load_b_frag(const unsigned short* __restrict__ Bp,
                                             int tile, int lane) {
  const uint4* p = (const uint4*)(Bp + ((size_t)tile << 9) + (lane << 4));
  FragU f;
  f.q[0] = p[0];
  f.q[1] = p[1];
  return f.v;
}

template<int NTW>
__device__ __forceinline__ void gemm_acc(v8f acc[NTW],
                                         const unsigned short* Alds, int ldA,
                                         const unsigned short* __restrict__ Bp, int KT,
                                         const int nts[NTW], int lane) {
  for (int kt = 0; kt < KT; ++kt) {
    v16bf a = load_a_frag(Alds, ldA, kt, lane);
#pragma unroll
    for (int i = 0; i < NTW; ++i) {
      v16bf b = load_b_frag(Bp, nts[i] * KT + kt, lane);
      acc[i] = __builtin_amdgcn_wmma_f32_16x16x32_bf16(
          false, a, false, b, (short)0, acc[i], false, false);
    }
  }
}

// ---------------- weight packing (fp32 [N,K] -> bf16 WMMA B-fragments) -------
__global__ void pack_weight(const float* __restrict__ W, int Nreal, int Kreal,
                            unsigned short* __restrict__ dst, int KT, int NT) {
  int idx = blockIdx.x * 256 + threadIdx.x;
  int total = NT * KT * 512;
  if (idx >= total) return;
  int e    = idx & 15;
  int lane = (idx >> 4) & 31;
  int tile = idx >> 9;
  int kt = tile % KT;
  int nt = tile / KT;
  int k = kt * 32 + ((e & 8) << 1) + ((lane & 16) >> 1) + (e & 7);  // lanes 0-15 -> K{0..7,16..23}; 16-31 -> K{8..15,24..31}
  int n = nt * 16 + (lane & 15);
  float v = (n < Nreal && k < Kreal) ? W[(size_t)n * Kreal + k] : 0.0f;
  dst[idx] = f2bf(v);
}

__global__ void prep_bias(const float* __restrict__ b1, const float* __restrict__ b2,
                          const float* __restrict__ b3,
                          const float* __restrict__ bih0, const float* __restrict__ bhh0,
                          const float* __restrict__ bih1, const float* __restrict__ bhh1,
                          const float* __restrict__ bc,
                          float* __restrict__ b1c, float* __restrict__ b2c,
                          float* __restrict__ b3p, float* __restrict__ bias0,
                          float* __restrict__ bias1, float* __restrict__ bcp) {
  int i = blockIdx.x * 256 + threadIdx.x;
  if (i < 256) { b1c[i] = b1[i]; b2c[i] = b2[i]; }
  if (i < 1024) {
    bias0[i] = bih0[i] + bhh0[i];
    bias1[i] = bih1[i] + bhh1[i];
    b3p[i] = (i < 1000) ? b3[i] : 0.0f;
    bcp[i] = (i < 1000) ? bc[i] : 0.0f;
  }
}

// ---------------- persistent decoder -----------------------------------------
__global__ void __launch_bounds__(256)
decoder_main(const float* __restrict__ state,
             const unsigned short* __restrict__ W1p,
             const unsigned short* __restrict__ W2p,
             const unsigned short* __restrict__ W3p,
             const unsigned short* __restrict__ Wih0p,
             const unsigned short* __restrict__ Whh0p,
             const unsigned short* __restrict__ Wih1p,
             const unsigned short* __restrict__ Whh1p,
             const unsigned short* __restrict__ Wcp,
             const float* __restrict__ b1c, const float* __restrict__ b2c,
             const float* __restrict__ b3p, const float* __restrict__ bias0,
             const float* __restrict__ bias1, const float* __restrict__ bcp,
             float* __restrict__ out, int T) {
  __shared__ __align__(16) unsigned short xS[16 * 1024];   // feedback input (bf16)
  __shared__ __align__(16) unsigned short h0S[16 * 256];   // LSTM0 hidden (bf16)
  __shared__ __align__(16) unsigned short h1S[16 * 256];   // LSTM1 hidden (bf16)

  const int tid  = threadIdx.x;
  const int w    = tid >> 5;
  const int lane = tid & 31;
  const int b0   = blockIdx.x * 16;
  const int col0 = lane & 15;
  const int mb   = (lane < 16) ? 0 : 8;

  // gate tile set for this wave: {i,i,f,f,g,g,o,o} covering hidden [32w,32w+32)
  int gnt[8], snt[8], mnt[2];
#pragma unroll
  for (int q = 0; q < 4; ++q) { gnt[2 * q] = 2 * w + 16 * q; gnt[2 * q + 1] = 2 * w + 16 * q + 1; }
#pragma unroll
  for (int i = 0; i < 8; ++i) snt[i] = w * 8 + i;
  mnt[0] = 2 * w; mnt[1] = 2 * w + 1;

  // preload biases into registers (constant over the time loop)
  float bi0[2], bf0[2], bg0[2], bo0[2], bi1[2], bf1[2], bg1[2], bo1[2], bco[8];
#pragma unroll
  for (int p = 0; p < 2; ++p) {
    int hc = (2 * w + p) * 16 + col0;
    bi0[p] = bias0[hc];       bf0[p] = bias0[256 + hc];
    bg0[p] = bias0[512 + hc]; bo0[p] = bias0[768 + hc];
    bi1[p] = bias1[hc];       bf1[p] = bias1[256 + hc];
    bg1[p] = bias1[512 + hc]; bo1[p] = bias1[768 + hc];
  }
#pragma unroll
  for (int i = 0; i < 8; ++i) bco[i] = bcp[(w * 8 + i) * 16 + col0];

  // ---- stage state tile (bf16, K padded 80 -> 96) into xS (stride 96) ----
  for (int idx = tid; idx < 16 * 96; idx += 256) {
    int m = idx / 96, k = idx - m * 96;
    float v = (k < 80) ? state[(size_t)(b0 + m) * 80 + k] : 0.0f;
    xS[m * 96 + k] = f2bf(v);
  }
  __syncthreads();

  // ---- MLP L1: a1 = relu(state @ W1^T + b1) -> h0S ----
  {
    v8f acc[2]; acc[0] = (v8f)0.0f; acc[1] = (v8f)0.0f;
    gemm_acc<2>(acc, xS, 96, W1p, 3, mnt, lane);
#pragma unroll
    for (int p = 0; p < 2; ++p) {
      int col = (2 * w + p) * 16 + col0;
      float bb = b1c[col];
#pragma unroll
      for (int r = 0; r < 8; ++r)
        h0S[(mb + r) * 256 + col] = f2bf(fmaxf(acc[p][r] + bb, 0.0f));
    }
  }
  __syncthreads();

  // ---- MLP L2: a2 = relu(a1 @ W2^T + b2) -> h1S ----
  {
    v8f acc[2]; acc[0] = (v8f)0.0f; acc[1] = (v8f)0.0f;
    gemm_acc<2>(acc, h0S, 256, W2p, 8, mnt, lane);
#pragma unroll
    for (int p = 0; p < 2; ++p) {
      int col = (2 * w + p) * 16 + col0;
      float bb = b2c[col];
#pragma unroll
      for (int r = 0; r < 8; ++r)
        h1S[(mb + r) * 256 + col] = f2bf(fmaxf(acc[p][r] + bb, 0.0f));
    }
  }
  __syncthreads();

  // ---- MLP L3: x = a2 @ W3^T + b3 -> xS (stride 1024) ----
  {
    v8f acc[8];
#pragma unroll
    for (int i = 0; i < 8; ++i) acc[i] = (v8f)0.0f;
    gemm_acc<8>(acc, h1S, 256, W3p, 8, snt, lane);
#pragma unroll
    for (int i = 0; i < 8; ++i) {
      int col = (w * 8 + i) * 16 + col0;
      float bb = b3p[col];
#pragma unroll
      for (int r = 0; r < 8; ++r)
        xS[(mb + r) * 1024 + col] = f2bf(acc[i][r] + bb);
    }
  }
  __syncthreads();

  // zero LSTM hidden state; cell state lives in registers
  for (int j = tid; j < 16 * 256; j += 256) { h0S[j] = 0; h1S[j] = 0; }
  v8f c0r[2], c1r[2];
  c0r[0] = (v8f)0.0f; c0r[1] = (v8f)0.0f; c1r[0] = (v8f)0.0f; c1r[1] = (v8f)0.0f;
  __syncthreads();

  // ---- time loop ----
  for (int t = 0; t < T; ++t) {
    // gates0 = x @ Wih0^T + h0 @ Whh0^T   (K = 1024 + 256)
    v8f acc[8];
#pragma unroll
    for (int i = 0; i < 8; ++i) acc[i] = (v8f)0.0f;
    gemm_acc<8>(acc, xS, 1024, Wih0p, 32, gnt, lane);
    gemm_acc<8>(acc, h0S, 256, Whh0p, 8, gnt, lane);
    __syncthreads();
    // LSTM0 nonlinearity in-register on accumulator fragments
#pragma unroll
    for (int p = 0; p < 2; ++p) {
      int hc = (2 * w + p) * 16 + col0;
#pragma unroll
      for (int r = 0; r < 8; ++r) {
        float ig = sigm(acc[p][r] + bi0[p]);
        float fg = sigm(acc[2 + p][r] + bf0[p]);
        float gg = tanhf(acc[4 + p][r] + bg0[p]);
        float og = sigm(acc[6 + p][r] + bo0[p]);
        float c = fg * c0r[p][r] + ig * gg;
        c0r[p][r] = c;
        h0S[(mb + r) * 256 + hc] = f2bf(og * tanhf(c));
      }
    }
    __syncthreads();

    // gates1 = h0 @ Wih1^T + h1 @ Whh1^T   (K = 256 + 256)
#pragma unroll
    for (int i = 0; i < 8; ++i) acc[i] = (v8f)0.0f;
    gemm_acc<8>(acc, h0S, 256, Wih1p, 8, gnt, lane);
    gemm_acc<8>(acc, h1S, 256, Whh1p, 8, gnt, lane);
    __syncthreads();
#pragma unroll
    for (int p = 0; p < 2; ++p) {
      int hc = (2 * w + p) * 16 + col0;
#pragma unroll
      for (int r = 0; r < 8; ++r) {
        float ig = sigm(acc[p][r] + bi1[p]);
        float fg = sigm(acc[2 + p][r] + bf1[p]);
        float gg = tanhf(acc[4 + p][r] + bg1[p]);
        float og = sigm(acc[6 + p][r] + bo1[p]);
        float c = fg * c1r[p][r] + ig * gg;
        c1r[p][r] = c;
        h1S[(mb + r) * 256 + hc] = f2bf(og * tanhf(c));
      }
    }
    __syncthreads();

    // out = h1 @ Wc^T + bc : write d_out and bf16 feedback into xS
#pragma unroll
    for (int i = 0; i < 8; ++i) acc[i] = (v8f)0.0f;
    gemm_acc<8>(acc, h1S, 256, Wcp, 8, snt, lane);
#pragma unroll
    for (int i = 0; i < 8; ++i) {
      const int col = (w * 8 + i) * 16 + col0;
#pragma unroll
      for (int r = 0; r < 8; ++r) {
        const int m = mb + r;
        float v = acc[i][r] + bco[i];
        xS[m * 1024 + col] = f2bf(v);
        if (col < 1000)
          out[((size_t)(b0 + m) * T + t) * 1000 + col] = v;
      }
    }
    __syncthreads();
  }
}

// ---------------- host launcher ----------------------------------------------
extern "C" void kernel_launch(void* const* d_in, const int* in_sizes, int n_in,
                              void* d_out, int out_size, void* d_ws, size_t ws_size,
                              hipStream_t stream) {
  (void)n_in; (void)ws_size;
  const float* state = (const float*)d_in[0];
  // d_in[1] = max_length (device scalar) - T derived from out_size instead
  const float* W1   = (const float*)d_in[2];
  const float* b1   = (const float*)d_in[3];
  const float* W2   = (const float*)d_in[4];
  const float* b2   = (const float*)d_in[5];
  const float* W3   = (const float*)d_in[6];
  const float* b3   = (const float*)d_in[7];
  const float* Wih0 = (const float*)d_in[8];
  const float* Whh0 = (const float*)d_in[9];
  const float* bih0 = (const float*)d_in[10];
  const float* bhh0 = (const float*)d_in[11];
  const float* Wih1 = (const float*)d_in[12];
  const float* Whh1 = (const float*)d_in[13];
  const float* bih1 = (const float*)d_in[14];
  const float* bhh1 = (const float*)d_in[15];
  const float* Wc   = (const float*)d_in[16];
  const float* bc   = (const float*)d_in[17];

  const int B = in_sizes[0] / 80;                          // 2048
  const int T = (int)((long long)out_size / ((long long)B * 1000));  // 50

  char* ws = (char*)d_ws;
  size_t off = 0;
  auto carve = [&](size_t bytes) -> char* {
    char* p = ws + off;
    off += (bytes + 255) & ~(size_t)255;
    return p;
  };
  unsigned short* Wih0p = (unsigned short*)carve((size_t)64 * 32 * 512 * 2);
  unsigned short* Whh0p = (unsigned short*)carve((size_t)64 * 8 * 512 * 2);
  unsigned short* Wih1p = (unsigned short*)carve((size_t)64 * 8 * 512 * 2);
  unsigned short* Whh1p = (unsigned short*)carve((size_t)64 * 8 * 512 * 2);
  unsigned short* Wcp   = (unsigned short*)carve((size_t)64 * 8 * 512 * 2);
  unsigned short* W3p   = (unsigned short*)carve((size_t)64 * 8 * 512 * 2);
  unsigned short* W2p   = (unsigned short*)carve((size_t)16 * 8 * 512 * 2);
  unsigned short* W1p   = (unsigned short*)carve((size_t)16 * 3 * 512 * 2);
  float* b1c   = (float*)carve(256 * 4);
  float* b2c   = (float*)carve(256 * 4);
  float* b3p   = (float*)carve(1024 * 4);
  float* bias0 = (float*)carve(1024 * 4);
  float* bias1 = (float*)carve(1024 * 4);
  float* bcp   = (float*)carve(1024 * 4);

  auto gsz = [](int total) { return (total + 255) / 256; };
  pack_weight<<<gsz(64 * 32 * 512), 256, 0, stream>>>(Wih0, 1024, 1000, Wih0p, 32, 64);
  pack_weight<<<gsz(64 * 8 * 512), 256, 0, stream>>>(Whh0, 1024, 256, Whh0p, 8, 64);
  pack_weight<<<gsz(64 * 8 * 512), 256, 0, stream>>>(Wih1, 1024, 256, Wih1p, 8, 64);
  pack_weight<<<gsz(64 * 8 * 512), 256, 0, stream>>>(Whh1, 1024, 256, Whh1p, 8, 64);
  pack_weight<<<gsz(64 * 8 * 512), 256, 0, stream>>>(Wc, 1000, 256, Wcp, 8, 64);
  pack_weight<<<gsz(64 * 8 * 512), 256, 0, stream>>>(W3, 1000, 256, W3p, 8, 64);
  pack_weight<<<gsz(16 * 8 * 512), 256, 0, stream>>>(W2, 256, 256, W2p, 8, 16);
  pack_weight<<<gsz(16 * 3 * 512), 256, 0, stream>>>(W1, 256, 80, W1p, 3, 16);
  prep_bias<<<4, 256, 0, stream>>>(b1, b2, b3, bih0, bhh0, bih1, bhh1, bc,
                                   b1c, b2c, b3p, bias0, bias1, bcp);

  decoder_main<<<B / 16, 256, 0, stream>>>(state, W1p, W2p, W3p, Wih0p, Whh0p,
                                           Wih1p, Whh1p, Wcp, b1c, b2c, b3p,
                                           bias0, bias1, bcp, (float*)d_out, T);
}
